// FastSelfAttention_8572754722986
// MI455X (gfx1250) — compile-verified, matrix operand order checked by reading, not run on hardware
//
#include <hip/hip_runtime.h>

// ---------------- types ----------------
typedef __attribute__((ext_vector_type(16))) __bf16 bf16x16;
typedef __attribute__((ext_vector_type(8)))  float  f32x8;
typedef __attribute__((address_space(3))) void lds_void;

#define B_  8
#define S_  4096
#define D_  768
#define H_  12
#define DH_ 64
#define NROWS (B_ * S_)           // 32768
#define INV_SQRT_DH 0.125f

// ---------------- helpers ----------------
__device__ __forceinline__ unsigned short f2bf(float f) {
    unsigned int x = __float_as_uint(f);
    unsigned int r = (x + 0x7FFFu + ((x >> 16) & 1u)) >> 16;   // RNE
    return (unsigned short)r;
}
__device__ __forceinline__ float bf2f(unsigned short u) {
    return __uint_as_float(((unsigned int)u) << 16);
}

// Async copy 16B (one wave-instruction: 32 lanes x 16B) global -> LDS.
__device__ __forceinline__ void async_copy16(void* ldsDst, const void* gSrc) {
    unsigned lds_off = (unsigned)(unsigned long long)(lds_void*)ldsDst;
    unsigned long long ga = (unsigned long long)gSrc;
    asm volatile("global_load_async_to_lds_b128 %0, %1, off"
                 :: "v"(lds_off), "v"(ga) : "memory");
}
__device__ __forceinline__ void wait_async0() {
    asm volatile("s_wait_asynccnt 0" ::: "memory");
}

// A/B fragment element order for v_wmma_*_16x16x32_bf16 (per lane):
//   slots 0..7  = K = kchunk+0..7     (kchunk = k0 + (lane>=16 ? 8 : 0))
//   slots 8..15 = K = kchunk+16..23
// p must already point at (row_base + kchunk); works for global or LDS pointers.
__device__ __forceinline__ bf16x16 frag_from_bf16(const unsigned short* p) {
    union { bf16x16 v; uint4 q[2]; } u;
    u.q[0] = *(const uint4*)(p);
    u.q[1] = *(const uint4*)(p + 16);
    return u.v;
}
__device__ __forceinline__ f32x8 wmma_bf16(bf16x16 a, bf16x16 b, f32x8 c) {
    return __builtin_amdgcn_wmma_f32_16x16x32_bf16(false, a, false, b, (short)0, c,
                                                   false, false);
}

// ---------------- kernels ----------------

// fp32 -> bf16 cast, scalar (weights)
__global__ __launch_bounds__(256) void cast_bf16_kernel(const float* __restrict__ in,
                                                        unsigned short* __restrict__ out,
                                                        int n) {
    int i = blockIdx.x * 256 + threadIdx.x;
    if (i < n) out[i] = f2bf(in[i]);
}

// fp32 -> bf16 cast, x4 vectorized (hidden_states). n4 = elems/4
__global__ __launch_bounds__(256) void cast4_bf16_kernel(const float* __restrict__ in,
                                                         unsigned short* __restrict__ out,
                                                         int n4) {
    int i = blockIdx.x * 256 + threadIdx.x;
    if (i < n4) {
        float4 f = ((const float4*)in)[i];
        union { unsigned short s[4]; uint2 v; } u;
        u.s[0] = f2bf(f.x); u.s[1] = f2bf(f.y);
        u.s[2] = f2bf(f.z); u.s[3] = f2bf(f.w);
        ((uint2*)out)[i] = u.v;
    }
}

// wv = bf16(pooled_k[b, k] * q[row, k]); x4 vectorized over NROWS*D elems
__global__ __launch_bounds__(256) void wv_kernel(const float* __restrict__ qf,
                                                 const float* __restrict__ pooled_k,
                                                 unsigned short* __restrict__ wvb) {
    int i = blockIdx.x * 256 + threadIdx.x;      // i in [0, NROWS*D/4)
    int e = i * 4;
    int k = e % D_;
    int b = e / (S_ * D_);
    float4 f = *(const float4*)(qf + e);
    const float* pk = pooled_k + b * D_ + k;
    union { unsigned short s[4]; uint2 v; } u;
    u.s[0] = f2bf(f.x * pk[0]); u.s[1] = f2bf(f.y * pk[1]);
    u.s[2] = f2bf(f.z * pk[2]); u.s[3] = f2bf(f.w * pk[3]);
    *(uint2*)(wvb + e) = u.v;
}

// Fused q/k projection GEMM: q = hs@Wq.T + bq (fp32 out), k = hs@Wk.T + bk (bf16 out)
// grid = (NROWS/128, D/64), block = 256 (8 waves, 4(M)x2(N) of 32x32 tiles)
// A tile (128x32 bf16, 8KB) staged in LDS via async copies, double buffered.
__global__ __launch_bounds__(256) void gemm_qk_kernel(
    const unsigned short* __restrict__ hsb, const unsigned short* __restrict__ Wqb,
    const unsigned short* __restrict__ Wkb, const float* __restrict__ bq,
    const float* __restrict__ bk, float* __restrict__ qf,
    unsigned short* __restrict__ kb) {
    __shared__ __align__(16) unsigned short aS[2][128][32];   // 16 KB

    const int tid  = threadIdx.x;
    const int lane = tid & 31;
    const int wave = tid >> 5;
    const int mw = wave & 3, nw = wave >> 2;
    const int l15 = lane & 15;
    const int lh8 = (lane >> 4) * 8;
    const int mTile = blockIdx.x * 128;
    const int mBase = mTile + mw * 32;
    const int nBase = blockIdx.y * 64 + nw * 32;

    // async chunk assignment: 512 x 16B chunks per k-step, 2 per thread
    const int c0 = tid * 2, c1 = c0 + 1;
    const int r0 = c0 >> 2, o0 = (c0 & 3) * 8;
    const int r1 = c1 >> 2, o1 = (c1 & 3) * 8;
    const unsigned short* g0 = hsb + (size_t)(mTile + r0) * D_ + o0;
    const unsigned short* g1 = hsb + (size_t)(mTile + r1) * D_ + o1;

    async_copy16(&aS[0][r0][o0], g0);
    async_copy16(&aS[0][r1][o1], g1);

    const unsigned short* bq0p = Wqb + (size_t)(nBase + l15) * D_ + lh8;
    const unsigned short* bq1p = bq0p + 16 * D_;
    const unsigned short* bk0p = Wkb + (size_t)(nBase + l15) * D_ + lh8;
    const unsigned short* bk1p = bk0p + 16 * D_;

    f32x8 cq[2][2] = {{{0}}};
    f32x8 ck[2][2] = {{{0}}};

    for (int it = 0; it < D_ / 32; ++it) {
        const int cur = it & 1;
        const int k0 = it * 32;
        wait_async0();          // own async stores to LDS complete
        __syncthreads();        // => whole tile visible to all waves
        if (it + 1 < D_ / 32) {
            const int kn = k0 + 32;
            async_copy16(&aS[cur ^ 1][r0][o0], g0 + kn);
            async_copy16(&aS[cur ^ 1][r1][o1], g1 + kn);
        }
        __builtin_prefetch(bq0p + k0 + 128, 0, 1);
        __builtin_prefetch(bk0p + k0 + 128, 0, 1);

        bf16x16 a0 = frag_from_bf16(&aS[cur][mw * 32 + l15][lh8]);
        bf16x16 a1 = frag_from_bf16(&aS[cur][mw * 32 + 16 + l15][lh8]);

        bf16x16 b0 = frag_from_bf16(bq0p + k0);
        bf16x16 b1 = frag_from_bf16(bq1p + k0);
        cq[0][0] = wmma_bf16(a0, b0, cq[0][0]);
        cq[1][0] = wmma_bf16(a1, b0, cq[1][0]);
        cq[0][1] = wmma_bf16(a0, b1, cq[0][1]);
        cq[1][1] = wmma_bf16(a1, b1, cq[1][1]);

        b0 = frag_from_bf16(bk0p + k0);
        b1 = frag_from_bf16(bk1p + k0);
        ck[0][0] = wmma_bf16(a0, b0, ck[0][0]);
        ck[1][0] = wmma_bf16(a1, b0, ck[1][0]);
        ck[0][1] = wmma_bf16(a0, b1, ck[0][1]);
        ck[1][1] = wmma_bf16(a1, b1, ck[1][1]);

        __syncthreads();        // protect cur buffer from next refill (WAR)
    }

#pragma unroll
    for (int j = 0; j < 2; ++j) {
        const int n = nBase + j * 16 + l15;
        const float bqv = bq[n];
        const float bkv = bk[n];
#pragma unroll
        for (int i = 0; i < 2; ++i) {
#pragma unroll
            for (int r = 0; r < 8; ++r) {
                const size_t m = (size_t)(mBase + i * 16 + lh8 + r);
                qf[m * D_ + n] = cq[i][j][r] + bqv;
                kb[m * D_ + n] = f2bf(ck[i][j][r] + bkv);
            }
        }
    }
}

// Final GEMM: out = wv @ Wt.T + bt + q, wv pre-packed bf16. Same tiling as above.
__global__ __launch_bounds__(256) void gemm_out_kernel(
    const unsigned short* __restrict__ wvb, const unsigned short* __restrict__ Wtb,
    const float* __restrict__ bt, const float* __restrict__ qf,
    float* __restrict__ out) {
    __shared__ __align__(16) unsigned short aS[2][128][32];

    const int tid  = threadIdx.x;
    const int lane = tid & 31;
    const int wave = tid >> 5;
    const int mw = wave & 3, nw = wave >> 2;
    const int l15 = lane & 15;
    const int lh8 = (lane >> 4) * 8;
    const int mTile = blockIdx.x * 128;
    const int mBase = mTile + mw * 32;
    const int nBase = blockIdx.y * 64 + nw * 32;

    const int c0 = tid * 2, c1 = c0 + 1;
    const int r0 = c0 >> 2, o0 = (c0 & 3) * 8;
    const int r1 = c1 >> 2, o1 = (c1 & 3) * 8;
    const unsigned short* g0 = wvb + (size_t)(mTile + r0) * D_ + o0;
    const unsigned short* g1 = wvb + (size_t)(mTile + r1) * D_ + o1;

    async_copy16(&aS[0][r0][o0], g0);
    async_copy16(&aS[0][r1][o1], g1);

    const unsigned short* b0p = Wtb + (size_t)(nBase + l15) * D_ + lh8;
    const unsigned short* b1p = b0p + 16 * D_;

    f32x8 c[2][2] = {{{0}}};
    for (int it = 0; it < D_ / 32; ++it) {
        const int cur = it & 1;
        const int k0 = it * 32;
        wait_async0();
        __syncthreads();
        if (it + 1 < D_ / 32) {
            const int kn = k0 + 32;
            async_copy16(&aS[cur ^ 1][r0][o0], g0 + kn);
            async_copy16(&aS[cur ^ 1][r1][o1], g1 + kn);
        }
        __builtin_prefetch(b0p + k0 + 128, 0, 1);

        bf16x16 a0 = frag_from_bf16(&aS[cur][mw * 32 + l15][lh8]);
        bf16x16 a1 = frag_from_bf16(&aS[cur][mw * 32 + 16 + l15][lh8]);
        bf16x16 b0 = frag_from_bf16(b0p + k0);
        bf16x16 b1 = frag_from_bf16(b1p + k0);
        c[0][0] = wmma_bf16(a0, b0, c[0][0]);
        c[1][0] = wmma_bf16(a1, b0, c[1][0]);
        c[0][1] = wmma_bf16(a0, b1, c[0][1]);
        c[1][1] = wmma_bf16(a1, b1, c[1][1]);

        __syncthreads();
    }

#pragma unroll
    for (int j = 0; j < 2; ++j) {
        const int n = nBase + j * 16 + l15;
        const float btv = bt[n];
#pragma unroll
        for (int i = 0; i < 2; ++i) {
#pragma unroll
            for (int r = 0; r < 8; ++r) {
                const size_t m = (size_t)(mBase + i * 16 + lh8 + r);
                out[m * D_ + n] = c[i][j][r] + btv + qf[m * D_ + n];
            }
        }
    }
}

// Per-row skinny GEMM: logits[b,h,s] = (q[b,s,:]·Wqa[h,:] + bqa[h])*inv_sqrt + mask[b,s]
__global__ __launch_bounds__(256) void qa_logits_kernel(
    const float* __restrict__ qf, const float* __restrict__ Wqa,
    const float* __restrict__ bqa, const float* __restrict__ mask,
    float* __restrict__ qa) {
    __shared__ float wqaS[H_ * D_];   // 36 KB
    for (int i = threadIdx.x; i < H_ * D_; i += 256) wqaS[i] = Wqa[i];
    __syncthreads();

    const int lane = threadIdx.x & 31;
    const int wave = threadIdx.x >> 5;
    const int row  = blockIdx.x * 8 + wave;
    const int b = row >> 12;
    const int s = row & (S_ - 1);
    const float* qr = qf + (size_t)row * D_;

    float acc[H_];
#pragma unroll
    for (int h = 0; h < H_; ++h) acc[h] = 0.0f;

    for (int t = 0; t < D_ / 32; ++t) {
        const int k = lane + t * 32;
        const float v = qr[k];
#pragma unroll
        for (int h = 0; h < H_; ++h) acc[h] += v * wqaS[h * D_ + k];
    }
#pragma unroll
    for (int h = 0; h < H_; ++h)
        for (int off = 16; off; off >>= 1) acc[h] += __shfl_xor(acc[h], off);

    if (lane == 0) {
        const float msk = mask[b * S_ + s];
#pragma unroll
        for (int h = 0; h < H_; ++h)
            qa[((size_t)(b * H_ + h)) * S_ + s] = (acc[h] + bqa[h]) * INV_SQRT_DH + msk;
    }
}

// Softmax over S + pooled_q[b,h,:] = sum_s w[s] * q[b,s,h*64:..]. grid = B*H
__global__ __launch_bounds__(256) void pool_q_kernel(const float* __restrict__ qa,
                                                     const float* __restrict__ qf,
                                                     float* __restrict__ pooled_q) {
    __shared__ float sm[S_];        // 16 KB
    __shared__ float redA[16];
    __shared__ float red2[256];
    const int bh = blockIdx.x;
    const int b = bh / H_, h = bh % H_;
    const float* lg = qa + (size_t)bh * S_;
    const int tid = threadIdx.x, lane = tid & 31, wave = tid >> 5;

    float mx = -1e30f;
    for (int s = tid; s < S_; s += 256) mx = fmaxf(mx, lg[s]);
    for (int off = 16; off; off >>= 1) mx = fmaxf(mx, __shfl_xor(mx, off));
    if (lane == 0) redA[wave] = mx;
    __syncthreads();
    mx = redA[0];
#pragma unroll
    for (int w = 1; w < 8; ++w) mx = fmaxf(mx, redA[w]);

    float sum = 0.0f;
    for (int s = tid; s < S_; s += 256) {
        float e = __expf(lg[s] - mx);
        sm[s] = e;
        sum += e;
    }
    for (int off = 16; off; off >>= 1) sum += __shfl_xor(sum, off);
    if (lane == 0) redA[8 + wave] = sum;
    __syncthreads();
    sum = 0.0f;
#pragma unroll
    for (int w = 0; w < 8; ++w) sum += redA[8 + w];
    const float inv = 1.0f / sum;

    const int d = tid & 63, g = tid >> 6;
    float acc = 0.0f;
    for (int s = g; s < S_; s += 4)
        acc += sm[s] * qf[((size_t)(b * S_ + s)) * D_ + h * DH_ + d];
    red2[tid] = acc;
    __syncthreads();
    if (g == 0)
        pooled_q[bh * DH_ + d] =
            (red2[d] + red2[64 + d] + red2[128 + d] + red2[192 + d]) * inv;
}

// Key attention: logits = (k·pooled_q)*inv + mask; softmax; pooled_k. grid = B*H
__global__ __launch_bounds__(256) void pool_k_kernel(const unsigned short* __restrict__ kb,
                                                     const float* __restrict__ pooled_q,
                                                     const float* __restrict__ mask,
                                                     float* __restrict__ pooled_k) {
    __shared__ float sm[S_];
    __shared__ float pq[DH_];
    __shared__ float redA[16];
    __shared__ float red2[256];
    const int bh = blockIdx.x;
    const int b = bh / H_, h = bh % H_;
    const int tid = threadIdx.x, lane = tid & 31, wave = tid >> 5;
    if (tid < DH_) pq[tid] = pooled_q[bh * DH_ + tid];
    __syncthreads();

    float mx = -1e30f;
    for (int s = tid; s < S_; s += 256) {
        const unsigned short* kr = kb + ((size_t)(b * S_ + s)) * D_ + h * DH_;
        float dot = 0.0f;
#pragma unroll
        for (int d0 = 0; d0 < DH_; d0 += 8) {
            union { uint4 q; unsigned short s8[8]; } u;
            u.q = *(const uint4*)(kr + d0);
#pragma unroll
            for (int e = 0; e < 8; ++e) dot += bf2f(u.s8[e]) * pq[d0 + e];
        }
        const float lgt = dot * INV_SQRT_DH + mask[b * S_ + s];
        sm[s] = lgt;
        mx = fmaxf(mx, lgt);
    }
    for (int off = 16; off; off >>= 1) mx = fmaxf(mx, __shfl_xor(mx, off));
    if (lane == 0) redA[wave] = mx;
    __syncthreads();
    mx = redA[0];
#pragma unroll
    for (int w = 1; w < 8; ++w) mx = fmaxf(mx, redA[w]);

    float sum = 0.0f;
    for (int s = tid; s < S_; s += 256) {
        float e = __expf(sm[s] - mx);
        sm[s] = e;
        sum += e;
    }
    for (int off = 16; off; off >>= 1) sum += __shfl_xor(sum, off);
    if (lane == 0) redA[8 + wave] = sum;
    __syncthreads();
    sum = 0.0f;
#pragma unroll
    for (int w = 0; w < 8; ++w) sum += redA[8 + w];
    const float inv = 1.0f / sum;

    const int d = tid & 63, g = tid >> 6;
    float acc = 0.0f;
    for (int s = g; s < S_; s += 4)
        acc += sm[s] * bf2f(kb[((size_t)(b * S_ + s)) * D_ + h * DH_ + d]);
    red2[tid] = acc;
    __syncthreads();
    if (g == 0)
        pooled_k[bh * DH_ + d] =
            (red2[d] + red2[64 + d] + red2[128 + d] + red2[192 + d]) * inv;
}

// ---------------- launcher ----------------
extern "C" void kernel_launch(void* const* d_in, const int* in_sizes, int n_in,
                              void* d_out, int out_size, void* d_ws, size_t ws_size,
                              hipStream_t stream) {
    const float* hs   = (const float*)d_in[0];
    const float* mask = (const float*)d_in[1];
    const float* Wq   = (const float*)d_in[2];
    const float* bq   = (const float*)d_in[3];
    const float* Wqa  = (const float*)d_in[4];
    const float* bqa  = (const float*)d_in[5];
    const float* Wk   = (const float*)d_in[6];
    const float* bk   = (const float*)d_in[7];
    const float* Wt   = (const float*)d_in[8];
    const float* bt   = (const float*)d_in[9];
    float* out = (float*)d_out;

    char* w = (char*)d_ws;
    size_t off = 0;
    float* qf = (float*)(w + off);                    off += (size_t)NROWS * D_ * 4; // 100.7 MB
    unsigned short* kb = (unsigned short*)(w + off);  off += (size_t)NROWS * D_ * 2; // 50.3 MB (reused as wvb)
    unsigned short* hsb = (unsigned short*)(w + off); off += (size_t)NROWS * D_ * 2; // 50.3 MB
    unsigned short* Wqb = (unsigned short*)(w + off); off += (size_t)D_ * D_ * 2;
    unsigned short* Wkb = (unsigned short*)(w + off); off += (size_t)D_ * D_ * 2;
    unsigned short* Wtb = (unsigned short*)(w + off); off += (size_t)D_ * D_ * 2;
    float* qa = (float*)(w + off);                    off += (size_t)B_ * H_ * S_ * 4;
    float* pooled_q = (float*)(w + off);              off += (size_t)B_ * H_ * DH_ * 4;
    float* pooled_k = (float*)(w + off);              off += (size_t)B_ * H_ * DH_ * 4;
    unsigned short* wvb = kb;   // kb is dead after pool_k; reuse for wv (same size)

    const int NW = D_ * D_;
    cast_bf16_kernel<<<(NW + 255) / 256, 256, 0, stream>>>(Wq, Wqb, NW);
    cast_bf16_kernel<<<(NW + 255) / 256, 256, 0, stream>>>(Wk, Wkb, NW);
    cast_bf16_kernel<<<(NW + 255) / 256, 256, 0, stream>>>(Wt, Wtb, NW);

    const int NH4 = NROWS * D_ / 4;
    cast4_bf16_kernel<<<NH4 / 256, 256, 0, stream>>>(hs, hsb, NH4);

    dim3 gGemm(NROWS / 128, D_ / 64);
    gemm_qk_kernel<<<gGemm, 256, 0, stream>>>(hsb, Wqb, Wkb, bq, bk, qf, kb);
    qa_logits_kernel<<<NROWS / 8, 256, 0, stream>>>(qf, Wqa, bqa, mask, qa);
    pool_q_kernel<<<B_ * H_, 256, 0, stream>>>(qa, qf, pooled_q);
    pool_k_kernel<<<B_ * H_, 256, 0, stream>>>(kb, pooled_q, mask, pooled_k);
    wv_kernel<<<NH4 / 256, 256, 0, stream>>>(qf, pooled_k, wvb);
    gemm_out_kernel<<<gGemm, 256, 0, stream>>>(wvb, Wtb, bt, qf, out);
}
